// EdgeConv_2980707303532
// MI455X (gfx1250) — matrix operand order repaired and checked by reading, not compiled, not run on hardware
//
#include <hip/hip_runtime.h>

#define LQ     2048
#define NB     4
#define CTX    4
#define KNN    20
#define CH_ACT 1344
#define NEDGE  (LQ * KNN)   // 40960 edge columns per batch

typedef _Float16 v8h  __attribute__((ext_vector_type(8)));
typedef _Float16 v16h __attribute__((ext_vector_type(16)));
typedef float    v8f  __attribute__((ext_vector_type(8)));

// ---------------------------------------------------------------------------
// Register-blocked WMMA GEMM:  D[n][m] = sum_k A[m][k] * B[n][k]
// (B stored column-major: each column n has its K entries contiguous).
// One wave computes MB vertically-adjacent 16x16 D tiles for one N tile,
// loading each B fragment ONCE and reusing it across MB WMMAs (4x less
// L2 traffic on the B side; ~410 FLOP/B per K-step at MB=4).
// mode 0: plain store (Gram)
// mode 1: BN (g,b,mean,var per row m) + LeakyReLU(0.2)
// mode 2: + bias[m]
// ---------------------------------------------------------------------------
template <int MB>
__global__ __launch_bounds__(256)
void wmma_gemm_kernel(const _Float16* __restrict__ A, int lda,
                      const _Float16* __restrict__ Bm, int ldb,
                      float* __restrict__ D, int ldd,
                      int nGroupsM, int nTilesN, int Ktot, int mode,
                      const float* __restrict__ p0, const float* __restrict__ p1,
                      const float* __restrict__ p2, const float* __restrict__ p3)
{
    int wave = blockIdx.x * (blockDim.x >> 5) + (threadIdx.x >> 5);
    if (wave >= nGroupsM * nTilesN) return;         // uniform per wave: EXEC stays all-ones
    int mg = wave % nGroupsM;                       // M fastest: waves in a block share nt
    int nt = wave / nGroupsM;
    int lane = threadIdx.x & 31;
    int half = lane >> 4;                           // lane half selects K sub-block
    int lm   = lane & 15;

    const _Float16* arow[MB];
#pragma unroll
    for (int i = 0; i < MB; ++i)
        arow[i] = A + (size_t)((mg * MB + i) * 16 + lm) * lda;   // A row for this lane
    const _Float16* bcol = Bm + (size_t)(nt * 16 + lm) * ldb;    // B column for this lane

    v8f acc[MB];
#pragma unroll
    for (int i = 0; i < MB; ++i) acc[i] = (v8f){};

    for (int kb = 0; kb < Ktot; kb += 32) {
        __builtin_prefetch(bcol + kb + 256, 0, 0);               // global_prefetch_b8
        // B 32x16 f16 fragment: lanes0-15 hold K[0..15], lanes16-31 hold K[16..31]
        v16h bf = *(const v16h*)(bcol + kb + 16 * half);
#pragma unroll
        for (int i = 0; i < MB; ++i) {
            // A 16x32 f16 fragment: lanes0-15 hold K[0..15], lanes16-31 hold K[16..31]
            v8h a0 = *(const v8h*)(arow[i] + kb + 16 * half);
            v8h a1 = *(const v8h*)(arow[i] + kb + 16 * half + 8);
            v16h af;
#pragma unroll
            for (int j = 0; j < 8; ++j) { af[j] = a0[j]; af[8 + j] = a1[j]; }
            acc[i] = __builtin_amdgcn_wmma_f32_16x16x32_f16(false, af, false, bf,
                                                            (short)0, acc[i], false, false);
        }
    }

    float* drow = D + (size_t)(nt * 16 + lm) * ldd;
#pragma unroll
    for (int i = 0; i < MB; ++i) {
        int obase = (mg * MB + i) * 16 + half * 8;  // 8 contiguous output rows per lane
        v8f a = acc[i];
        if (mode == 1) {
#pragma unroll
            for (int r = 0; r < 8; ++r) {
                int o = obase + r;
                float sc = p0[o] * rsqrtf(p3[o] + 1e-5f);
                float y  = (a[r] - p2[o]) * sc + p1[o];
                a[r] = y >= 0.0f ? y : 0.2f * y;
            }
        } else if (mode == 2) {
#pragma unroll
            for (int r = 0; r < 8; ++r) a[r] += p0[obase + r];
        }
        *(v8f*)(drow + obase) = a;
    }
}

// ---------------------------------------------------------------------------
// Embedding gather -> act[b][l][32..64)=emb32, [192..320)=emb128, [832..1344)=emb512
// ---------------------------------------------------------------------------
__global__ void embed_kernel(const int* __restrict__ occ, const int* __restrict__ lev,
                             const int* __restrict__ octn,
                             const float* __restrict__ E0a, const float* __restrict__ E1a,
                             const float* __restrict__ E2a,
                             const float* __restrict__ E0b, const float* __restrict__ E1b,
                             const float* __restrict__ E2b,
                             const float* __restrict__ E0c, const float* __restrict__ E1c,
                             const float* __restrict__ E2c,
                             _Float16* __restrict__ act)
{
    int t = blockIdx.x * blockDim.x + threadIdx.x;
    if (t >= LQ * NB * CTX) return;
    int ctx = t % CTX; int r = t / CTX; int b = r % NB; int l = r / NB;
    size_t ii = ((size_t)l * NB + b) * CTX + ctx;
    int io = occ[ii], il = lev[ii], ic = octn[ii];
    _Float16* dst = act + ((size_t)b * LQ + l) * CH_ACT;

    int o32 = 32 + ctx * 8;
    for (int j = 0; j < 6; ++j) dst[o32 + j] = (_Float16)E0a[io * 6 + j];
    dst[o32 + 6] = (_Float16)E1a[il];
    dst[o32 + 7] = (_Float16)E2a[ic];

    int o128 = 192 + ctx * 32;
    for (int j = 0; j < 30; ++j) dst[o128 + j] = (_Float16)E0b[io * 30 + j];
    dst[o128 + 30] = (_Float16)E1b[il];
    dst[o128 + 31] = (_Float16)E2b[ic];

    int o512 = 832 + ctx * 128;
    for (int j = 0; j < 126; ++j) dst[o512 + j] = (_Float16)E0c[io * 126 + j];
    dst[o512 + 126] = (_Float16)E1c[il];
    dst[o512 + 127] = (_Float16)E2c[ic];
}

// ---------------------------------------------------------------------------
// Top-k helper (descending, stable: equal keys keep earlier index first)
// ---------------------------------------------------------------------------
__device__ __forceinline__ void topk_insert(float* key, int* kid, float p, int m)
{
    if (p <= key[KNN - 1]) return;
    int j = KNN - 1;
    while (j > 0 && key[j - 1] < p) { key[j] = key[j - 1]; kid[j] = kid[j - 1]; --j; }
    key[j] = p; kid[j] = m;
}

// kNN on raw positions (C=3) for layer 1
__global__ void knn3_kernel(const float* __restrict__ pos, int* __restrict__ idx, int b)
{
    int l = blockIdx.x * blockDim.x + threadIdx.x;
    if (l >= LQ) return;
    float cx = pos[((size_t)l * NB + b) * 3 + 0];
    float cy = pos[((size_t)l * NB + b) * 3 + 1];
    float cz = pos[((size_t)l * NB + b) * 3 + 2];
    float key[KNN]; int kid[KNN];
    for (int i = 0; i < KNN; ++i) { key[i] = -3.4e38f; kid[i] = 0; }
    for (int m = 0; m < LQ; ++m) {
        float dx = pos[((size_t)m * NB + b) * 3 + 0] - cx;
        float dy = pos[((size_t)m * NB + b) * 3 + 1] - cy;
        float dz = pos[((size_t)m * NB + b) * 3 + 2] - cz;
        topk_insert(key, kid, -(dx * dx + dy * dy + dz * dz), m);
    }
    for (int i = 0; i < KNN; ++i) idx[l * KNN + i] = kid[i];
}

__global__ void diag_kernel(const float* __restrict__ S, float* __restrict__ dg)
{
    int m = blockIdx.x * blockDim.x + threadIdx.x;
    if (m < LQ) dg[m] = S[(size_t)m * LQ + m];
}

// kNN from Gram matrix: pd = 2*S[l,m] - S[l,l] - S[m,m]
__global__ void topk_kernel(const float* __restrict__ S, const float* __restrict__ dg,
                            int* __restrict__ idx)
{
    int l = blockIdx.x * blockDim.x + threadIdx.x;
    if (l >= LQ) return;
    const float* row = S + (size_t)l * LQ;
    float dl = dg[l];
    float key[KNN]; int kid[KNN];
    for (int i = 0; i < KNN; ++i) { key[i] = -3.4e38f; kid[i] = 0; }
    for (int m = 0; m < LQ; ++m)
        topk_insert(key, kid, 2.0f * row[m] - dl - dg[m], m);
    for (int i = 0; i < KNN; ++i) idx[l * KNN + i] = kid[i];
}

// ---------------------------------------------------------------------------
// Edge feature builders: F column = [x_nbr - x_ctr ; x_ctr], K-contiguous f16
// ---------------------------------------------------------------------------
__global__ void build_feat1_kernel(const float* __restrict__ pos, const int* __restrict__ idx,
                                   _Float16* __restrict__ F, int b)
{
    int t = blockIdx.x * blockDim.x + threadIdx.x;
    if (t >= NEDGE) return;
    int l = t / KNN;
    int nb = idx[t];
    const float* pc = pos + ((size_t)l  * NB + b) * 3;
    const float* pn = pos + ((size_t)nb * NB + b) * 3;
    _Float16* dst = F + (size_t)t * 32;            // padded to K=32 with zeros
    for (int c = 0; c < 3; ++c) {
        dst[c]     = (_Float16)(pn[c] - pc[c]);
        dst[3 + c] = (_Float16)pc[c];
    }
    for (int c = 6; c < 32; ++c) dst[c] = (_Float16)0.0f;
}

__global__ void build_feat_kernel(const _Float16* __restrict__ act, const int* __restrict__ idx,
                                  _Float16* __restrict__ F, int b, int coff, int C)
{
    int t = blockIdx.x * blockDim.x + threadIdx.x;
    if (t >= NEDGE * C) return;
    int c = t % C; int e = t / C;
    int l = e / KNN;
    int nb = idx[e];
    float ctr = (float)act[((size_t)b * LQ + l)  * CH_ACT + coff + c];
    float nbv = (float)act[((size_t)b * LQ + nb) * CH_ACT + coff + c];
    F[(size_t)e * (2 * C) + c]     = (_Float16)(nbv - ctr);
    F[(size_t)e * (2 * C) + C + c] = (_Float16)ctr;
}

// max over k=20 edge columns -> act[b][l][coff + o] (f16)
__global__ void maxk_store_kernel(const float* __restrict__ conv, _Float16* __restrict__ act,
                                  int b, int coff, int O)
{
    int t = blockIdx.x * blockDim.x + threadIdx.x;
    if (t >= LQ * O) return;
    int o = t % O; int l = t / O;
    float mx = -3.4e38f;
    for (int kk = 0; kk < KNN; ++kk)
        mx = fmaxf(mx, conv[(size_t)(l * KNN + kk) * O + o]);
    act[((size_t)b * LQ + l) * CH_ACT + coff + o] = (_Float16)mx;
}

// mean & max over L of xc (xcf layout [b*L + l][512])
__global__ void reduce_xc_kernel(const float* __restrict__ xcf,
                                 float* __restrict__ avg, float* __restrict__ mxv)
{
    int t = blockIdx.x * blockDim.x + threadIdx.x;
    if (t >= NB * 512) return;
    int b = t / 512; int o = t % 512;
    float s = 0.0f, mx = -3.4e38f;
    for (int l = 0; l < LQ; ++l) {
        float v = xcf[((size_t)b * LQ + l) * 512 + o];
        s += v; mx = fmaxf(mx, v);
    }
    avg[t] = s / (float)LQ;
    mxv[t] = mx;
}

// final feature columns: [x5(1024) | avg(512) | max(512)], column n = l*B + b
__global__ void build_ff_kernel(const _Float16* __restrict__ act, const float* __restrict__ avg,
                                const float* __restrict__ mxv, _Float16* __restrict__ ff)
{
    int t = blockIdx.x * blockDim.x + threadIdx.x;
    if (t >= NB * LQ * 2048) return;
    int c = t & 2047; int n = t >> 11;
    int b = n % NB; int l = n / NB;
    _Float16 v;
    if (c < 1024)       v = act[((size_t)b * LQ + l) * CH_ACT + 320 + c];
    else if (c < 1536)  v = (_Float16)avg[b * 512 + (c - 1024)];
    else                v = (_Float16)mxv[b * 512 + (c - 1536)];
    ff[(size_t)n * 2048 + c] = v;
}

__global__ void cvt_f16_kernel(const float* __restrict__ s, _Float16* __restrict__ d, int n)
{
    int t = blockIdx.x * blockDim.x + threadIdx.x;
    if (t < n) d[t] = (_Float16)s[t];
}

// W1 (32x6) -> zero-padded (32x32) f16
__global__ void pad_w1_kernel(const float* __restrict__ s, _Float16* __restrict__ d)
{
    int t = blockIdx.x * blockDim.x + threadIdx.x;
    if (t >= 32 * 32) return;
    int o = t >> 5, c = t & 31;
    d[t] = (c < 6) ? (_Float16)s[o * 6 + c] : (_Float16)0.0f;
}

// ---------------------------------------------------------------------------
extern "C" void kernel_launch(void* const* d_in, const int* in_sizes, int n_in,
                              void* d_out, int out_size, void* d_ws, size_t ws_size,
                              hipStream_t stream)
{
    (void)in_sizes; (void)n_in; (void)out_size; (void)ws_size;
    const int*   occ   = (const int*)d_in[0];
    const int*   lev   = (const int*)d_in[1];
    const int*   octn  = (const int*)d_in[2];
    const float* pos   = (const float*)d_in[3];
    const float* E0a   = (const float*)d_in[4];
    const float* E1a   = (const float*)d_in[5];
    const float* E2a   = (const float*)d_in[6];
    const float* E0b   = (const float*)d_in[7];
    const float* E1b   = (const float*)d_in[8];
    const float* E2b   = (const float*)d_in[9];
    const float* E0c   = (const float*)d_in[10];
    const float* E1c   = (const float*)d_in[11];
    const float* E2c   = (const float*)d_in[12];
    const float* W1 = (const float*)d_in[13];
    const float* g1 = (const float*)d_in[14]; const float* b1 = (const float*)d_in[15];
    const float* m1 = (const float*)d_in[16]; const float* v1 = (const float*)d_in[17];
    const float* W3 = (const float*)d_in[18];
    const float* g3 = (const float*)d_in[19]; const float* b3 = (const float*)d_in[20];
    const float* m3 = (const float*)d_in[21]; const float* v3 = (const float*)d_in[22];
    const float* W5 = (const float*)d_in[23];
    const float* g5 = (const float*)d_in[24]; const float* b5 = (const float*)d_in[25];
    const float* m5 = (const float*)d_in[26]; const float* v5 = (const float*)d_in[27];
    const float* Wf = (const float*)d_in[28];
    const float* gf = (const float*)d_in[29]; const float* bf = (const float*)d_in[30];
    const float* mf = (const float*)d_in[31]; const float* vf = (const float*)d_in[32];
    const float* Wm = (const float*)d_in[33];
    const float* bm = (const float*)d_in[34];
    float* out = (float*)d_out;

    // ---- workspace carve-out (~220 MB) ----
    char* p = (char*)d_ws;
    auto alloc = [&](size_t bytes) -> char* {
        char* r = p; p += (bytes + 255) & ~(size_t)255; return r;
    };
    _Float16* act  = (_Float16*)alloc((size_t)NB * LQ * CH_ACT * 2);
    _Float16* W1h  = (_Float16*)alloc(32 * 32 * 2);
    _Float16* W3h  = (_Float16*)alloc(128 * 128 * 2);
    _Float16* W5h  = (_Float16*)alloc((size_t)512 * 512 * 2);
    _Float16* Wfh  = (_Float16*)alloc((size_t)512 * 1344 * 2);
    _Float16* Wmh  = (_Float16*)alloc((size_t)512 * 2048 * 2);
    float*    S    = (float*)alloc((size_t)LQ * LQ * 4);
    float*    dg   = (float*)alloc(LQ * 4);
    int*      idx  = (int*)alloc(LQ * KNN * 4);
    _Float16* F    = (_Float16*)alloc((size_t)NEDGE * 512 * 2);
    float*    conv = (float*)alloc((size_t)NEDGE * 512 * 4);
    float*    xcf  = (float*)alloc((size_t)NB * LQ * 512 * 4);
    float*    avg  = (float*)alloc(NB * 512 * 4);
    float*    mxv  = (float*)alloc(NB * 512 * 4);
    _Float16* ff   = (_Float16*)alloc((size_t)NB * LQ * 2048 * 2);

    // Dispatch: MB=4 when the M-tile count allows it (all GEMMs except the
    // tiny O=32 layer-1 conv), otherwise MB=2.
    auto gemm = [&](const _Float16* A, int lda, const _Float16* Bmt, int ldb,
                    float* D, int ldd, int O, int N, int Ktot, int mode,
                    const float* q0, const float* q1, const float* q2, const float* q3) {
        int tm = O / 16, tn = N / 16;
        if (tm % 4 == 0) {
            int groups = (tm / 4) * tn;
            wmma_gemm_kernel<4><<<(groups + 7) / 8, 256, 0, stream>>>(
                A, lda, Bmt, ldb, D, ldd, tm / 4, tn, Ktot, mode, q0, q1, q2, q3);
        } else {
            int groups = (tm / 2) * tn;
            wmma_gemm_kernel<2><<<(groups + 7) / 8, 256, 0, stream>>>(
                A, lda, Bmt, ldb, D, ldd, tm / 2, tn, Ktot, mode, q0, q1, q2, q3);
        }
    };

    // weight conversions
    pad_w1_kernel<<<4, 256, 0, stream>>>(W1, W1h);
    cvt_f16_kernel<<<(128 * 128 + 255) / 256, 256, 0, stream>>>(W3, W3h, 128 * 128);
    cvt_f16_kernel<<<(512 * 512 + 255) / 256, 256, 0, stream>>>(W5, W5h, 512 * 512);
    cvt_f16_kernel<<<(512 * 1344 + 255) / 256, 256, 0, stream>>>(Wf, Wfh, 512 * 1344);
    cvt_f16_kernel<<<(512 * 2048 + 255) / 256, 256, 0, stream>>>(Wm, Wmh, 512 * 2048);

    // embeddings into act
    embed_kernel<<<(LQ * NB * CTX + 255) / 256, 256, 0, stream>>>(
        occ, lev, octn, E0a, E1a, E2a, E0b, E1b, E2b, E0c, E1c, E2c, act);

    for (int b = 0; b < NB; ++b) {
        const _Float16* actb = act + (size_t)b * LQ * CH_ACT;
        // ---- layer 1 (C=3, direct kNN, K padded to 32) ----
        knn3_kernel<<<LQ / 256, 256, 0, stream>>>(pos, idx, b);
        build_feat1_kernel<<<(NEDGE + 255) / 256, 256, 0, stream>>>(pos, idx, F, b);
        gemm(W1h, 32, F, 32, conv, 32, 32, NEDGE, 32, 1, g1, b1, m1, v1);
        maxk_store_kernel<<<(LQ * 32 + 255) / 256, 256, 0, stream>>>(conv, act, b, 0, 32);
        // ---- layer 3 (C=64) ----
        gemm(actb, CH_ACT, actb, CH_ACT, S, LQ, LQ, LQ, 64, 0, nullptr, nullptr, nullptr, nullptr);
        diag_kernel<<<LQ / 256, 256, 0, stream>>>(S, dg);
        topk_kernel<<<LQ / 256, 256, 0, stream>>>(S, dg, idx);
        build_feat_kernel<<<(NEDGE * 64 + 255) / 256, 256, 0, stream>>>(act, idx, F, b, 0, 64);
        gemm(W3h, 128, F, 128, conv, 128, 128, NEDGE, 128, 1, g3, b3, m3, v3);
        maxk_store_kernel<<<(LQ * 128 + 255) / 256, 256, 0, stream>>>(conv, act, b, 64, 128);
        // ---- layer 5 (C=256) ----
        gemm(actb + 64, CH_ACT, actb + 64, CH_ACT, S, LQ, LQ, LQ, 256, 0,
             nullptr, nullptr, nullptr, nullptr);
        diag_kernel<<<LQ / 256, 256, 0, stream>>>(S, dg);
        topk_kernel<<<LQ / 256, 256, 0, stream>>>(S, dg, idx);
        build_feat_kernel<<<(NEDGE * 256 + 255) / 256, 256, 0, stream>>>(act, idx, F, b, 64, 256);
        gemm(W5h, 512, F, 512, conv, 512, 512, NEDGE, 512, 1, g5, b5, m5, v5);
        maxk_store_kernel<<<(LQ * 512 + 255) / 256, 256, 0, stream>>>(conv, act, b, 320, 512);
    }

    // fuse conv: xc (1344 -> 512), BN + leaky, columns n = b*L + l
    gemm(Wfh, CH_ACT, act, CH_ACT, xcf, 512, 512, NB * LQ, CH_ACT, 1, gf, bf, mf, vf);
    reduce_xc_kernel<<<(NB * 512 + 255) / 256, 256, 0, stream>>>(xcf, avg, mxv);
    build_ff_kernel<<<(NB * LQ * 2048 + 255) / 256, 256, 0, stream>>>(act, avg, mxv, ff);
    // final linear straight into d_out: column n = l*B + b -> (L,B,512) row-major
    gemm(Wmh, 2048, ff, 2048, out, 512, 512, NB * LQ, 2048, 2, bm, nullptr, nullptr, nullptr);
}
